// GCN_23862838297294
// MI455X (gfx1250) — compile-verified, matrix operand order checked by reading, not run on hardware
//
#include <hip/hip_runtime.h>
#include <cstdint>
#include <cstddef>

// ---------------------------------------------------------------------------
// Problem constants (match reference)
// ---------------------------------------------------------------------------
static constexpr int BG    = 16384;        // graphs
static constexpr int NODES = 11;
static constexpr int NN    = BG * NODES;   // 180224 total nodes (multiple of 16)
static constexpr int IND   = 256;
static constexpr int HH    = 256;
static constexpr int G4    = 4 * HH;       // 1024 gates
static constexpr int EPG   = NODES * 8;    // 88 edges per graph, contiguous

#define DEVFN __device__ __forceinline__

typedef __attribute__((ext_vector_type(16))) __bf16       v16bf;
typedef __attribute__((ext_vector_type(8)))  float        v8f;
typedef __attribute__((ext_vector_type(4)))  unsigned int u32x4;
typedef __attribute__((ext_vector_type(4)))  int          i32x4;

union FragBF { v16bf v; u32x4 q[2]; };

DEVFN unsigned short f2bf(float x) {
  unsigned int u = __float_as_uint(x);
  u += 0x7fffu + ((u >> 16) & 1u);          // round-to-nearest-even
  return (unsigned short)(u >> 16);
}
DEVFN float bf2f(unsigned short b) { return __uint_as_float(((unsigned int)b) << 16); }
DEVFN float sigf(float x) { return 1.0f / (1.0f + __expf(-x)); }

// CDNA5 async global->LDS path (guarded so compile cannot break)
#if __has_builtin(__builtin_amdgcn_global_load_async_to_lds_b128) && \
    __has_builtin(__builtin_amdgcn_s_wait_asynccnt)
#define HAVE_ASYNC_LDS 1
#else
#define HAVE_ASYNC_LDS 0
#endif
#define AS1 __attribute__((address_space(1)))
#define AS3 __attribute__((address_space(3)))

// ---------------------------------------------------------------------------
// Generic bf16 WMMA GEMM:  C[M,N] = A[M,K] @ B[K,N], B supplied transposed
// (Bt is [N][K] row-major bf16).  One wave computes a 16x64 C tile.
// Output either f32 (Cf) or bf16 (Cb) -- exactly one non-null.
// ---------------------------------------------------------------------------
__global__ __launch_bounds__(256)
void wmma_gemm_bf16(const unsigned short* __restrict__ A,
                    const unsigned short* __restrict__ Bt,
                    float* __restrict__ Cf,
                    unsigned short* __restrict__ Cb,
                    int M, int N, int K)
{
  const int lane   = threadIdx.x & 31;
  const int wave   = threadIdx.x >> 5;
  const int gw     = blockIdx.x * (blockDim.x >> 5) + wave;
  const int tilesN = N >> 6;                          // 64-wide tiles
  const int total  = (M >> 4) * tilesN;
  if (gw >= total) return;
  const int tileM  = gw / tilesN;
  const int tileN  = gw % tilesN;

  const int mrow  = lane & 15;
  const int khalf = lane >> 4;                        // K 0..7/16..23 vs 8..15/24..31

  const unsigned short* arow  = A  + (size_t)(tileM * 16 + mrow) * K;
  const unsigned short* brow0 = Bt + (size_t)(tileN * 64 + mrow) * K;

  v8f acc[4] = {};

  for (int kk = 0; kk < K; kk += 32) {
    __builtin_prefetch((const void*)(arow + kk + 128), 0, 1);   // global_prefetch_b8
    FragBF fa;
    fa.q[0] = *reinterpret_cast<const u32x4*>(arow + kk + khalf * 8);
    fa.q[1] = *reinterpret_cast<const u32x4*>(arow + kk + 16 + khalf * 8);
#pragma unroll
    for (int s = 0; s < 4; ++s) {
      const unsigned short* brow = brow0 + (size_t)(s * 16) * K;
      FragBF fb;
      fb.q[0] = *reinterpret_cast<const u32x4*>(brow + kk + khalf * 8);
      fb.q[1] = *reinterpret_cast<const u32x4*>(brow + kk + 16 + khalf * 8);
      acc[s] = __builtin_amdgcn_wmma_f32_16x16x32_bf16(
          false, fa.v, false, fb.v, (short)0, acc[s], false, false);
    }
  }

  const int r0 = tileM * 16 + khalf * 8;
  const int c0 = tileN * 64;
  if (Cb) {
#pragma unroll
    for (int s = 0; s < 4; ++s)
#pragma unroll
      for (int v = 0; v < 8; ++v)
        Cb[(size_t)(r0 + v) * N + c0 + s * 16 + mrow] = f2bf(acc[s][v]);
  } else {
#pragma unroll
    for (int s = 0; s < 4; ++s)
#pragma unroll
      for (int v = 0; v < 8; ++v)
        Cf[(size_t)(r0 + v) * N + c0 + s * 16 + mrow] = acc[s][v];
  }
}

// ---------------------------------------------------------------------------
// Fused LSTM step:  gates = [x_t | h] @ Wcomb^T  (+ nonlinearities, in-reg).
// Wave computes 16 rows x 16 j-columns; accumulator subtile s = gate s
// (B rows s*256 + j), so each lane holds i,f,g,o for its (row, j) pairs.
// ---------------------------------------------------------------------------
__global__ __launch_bounds__(256)
void lstm_step_wmma(const unsigned short* __restrict__ featbf, // [NN,256] bf16
                    const unsigned short* __restrict__ hin,    // [BG,256] bf16
                    unsigned short* __restrict__ hout,         // [BG,256] bf16
                    const unsigned short* __restrict__ Wt,     // [1024,512] bf16
                    const float* __restrict__ bc,              // bih+bhh [1024]
                    float* __restrict__ cbuf,                  // [BG,256] f32
                    float* __restrict__ hf32,                  // [BG,256] f32
                    int t)
{
  const int lane   = threadIdx.x & 31;
  const int wave   = threadIdx.x >> 5;
  const int gw     = blockIdx.x * (blockDim.x >> 5) + wave;
  const int tilesN = HH / 16;                    // 16 j-tiles
  if (gw >= (BG / 16) * tilesN) return;
  const int tileM = gw / tilesN;
  const int tileN = gw % tilesN;
  const int mrow  = lane & 15;
  const int khalf = lane >> 4;
  const int grow  = tileM * 16 + mrow;           // graph index for A loads

  const unsigned short* ax = featbf + ((size_t)(grow * NODES + t)) * IND;
  const unsigned short* ah = hin + (size_t)grow * HH;

  v8f acc[4] = {};

  // x part: K = 0..255 of Wcomb rows
  for (int kk = 0; kk < IND; kk += 32) {
    FragBF fa;
    fa.q[0] = *reinterpret_cast<const u32x4*>(ax + kk + khalf * 8);
    fa.q[1] = *reinterpret_cast<const u32x4*>(ax + kk + 16 + khalf * 8);
#pragma unroll
    for (int s = 0; s < 4; ++s) {
      const unsigned short* br = Wt + (size_t)(s * HH + tileN * 16 + mrow) * 512 + kk;
      FragBF fb;
      fb.q[0] = *reinterpret_cast<const u32x4*>(br + khalf * 8);
      fb.q[1] = *reinterpret_cast<const u32x4*>(br + 16 + khalf * 8);
      acc[s] = __builtin_amdgcn_wmma_f32_16x16x32_bf16(
          false, fa.v, false, fb.v, (short)0, acc[s], false, false);
    }
  }
  // h part: K = 256..511 of Wcomb rows
  for (int kk = 0; kk < HH; kk += 32) {
    FragBF fa;
    fa.q[0] = *reinterpret_cast<const u32x4*>(ah + kk + khalf * 8);
    fa.q[1] = *reinterpret_cast<const u32x4*>(ah + kk + 16 + khalf * 8);
#pragma unroll
    for (int s = 0; s < 4; ++s) {
      const unsigned short* br = Wt + (size_t)(s * HH + tileN * 16 + mrow) * 512 + 256 + kk;
      FragBF fb;
      fb.q[0] = *reinterpret_cast<const u32x4*>(br + khalf * 8);
      fb.q[1] = *reinterpret_cast<const u32x4*>(br + 16 + khalf * 8);
      acc[s] = __builtin_amdgcn_wmma_f32_16x16x32_bf16(
          false, fa.v, false, fb.v, (short)0, acc[s], false, false);
    }
  }

  // epilogue: gate nonlinearities fully in-register
  const int j = tileN * 16 + mrow;
  const float bi = bc[j], bf_ = bc[HH + j], bg = bc[2 * HH + j], bo = bc[3 * HH + j];
#pragma unroll
  for (int v = 0; v < 8; ++v) {
    const int row = tileM * 16 + v + 8 * khalf;
    const size_t idx = (size_t)row * HH + j;
    float gi = acc[0][v] + bi;
    float gf = acc[1][v] + bf_;
    float gg = acc[2][v] + bg;
    float go = acc[3][v] + bo;
    float c  = sigf(gf) * cbuf[idx] + sigf(gi) * tanhf(gg);
    float h  = sigf(go) * tanhf(c);
    cbuf[idx] = c;
    hf32[idx] = h;
    hout[idx] = f2bf(h);
  }
}

// ---------------------------------------------------------------------------
// Conversion / init helpers
// ---------------------------------------------------------------------------
__global__ void conv_f32_bf16(unsigned short* __restrict__ d,
                              const float* __restrict__ s, size_t n) {
  size_t i = (size_t)blockIdx.x * blockDim.x + threadIdx.x;
  if (i < n) d[i] = f2bf(s[i]);
}
// dst[n*K + k] = src[k*N + n]  (build B^T bf16 from row-major [K,N] f32)
__global__ void convT_f32_bf16(unsigned short* __restrict__ d,
                               const float* __restrict__ s, int K, int N) {
  size_t i = (size_t)blockIdx.x * blockDim.x + threadIdx.x;
  if (i >= (size_t)K * N) return;
  int n = (int)(i / K), k = (int)(i % K);
  d[i] = f2bf(s[(size_t)k * N + n]);
}
// pack rows of src [R,C] f32 into dst rows of stride S at column off (bf16)
__global__ void conv_rows_bf16(unsigned short* __restrict__ d,
                               const float* __restrict__ s,
                               int R, int C, int S, int off) {
  size_t i = (size_t)blockIdx.x * blockDim.x + threadIdx.x;
  if (i >= (size_t)R * C) return;
  int r = (int)(i / C), c = (int)(i % C);
  d[(size_t)r * S + off + c] = f2bf(s[i]);
}
__global__ void bias_comb_k(float* __restrict__ d, const float* __restrict__ a,
                            const float* __restrict__ b, int n) {
  int i = blockIdx.x * blockDim.x + threadIdx.x;
  if (i < n) d[i] = a[i] + b[i];
}
__global__ void zero_f32(float* p, size_t n) {
  size_t i = (size_t)blockIdx.x * blockDim.x + threadIdx.x;
  if (i < n) p[i] = 0.0f;
}
__global__ void zero_u16(unsigned short* p, size_t n) {
  size_t i = (size_t)blockIdx.x * blockDim.x + threadIdx.x;
  if (i < n) p[i] = 0;
}

// ---------------------------------------------------------------------------
// Per-graph normalized adjacency:  M[d][s] = dn[d] * cnt[d][s] * sn[s]
// ---------------------------------------------------------------------------
__global__ __launch_bounds__(128)
void graph_prep(const int* __restrict__ src, const int* __restrict__ dst,
                float* __restrict__ Mg)
{
  const int g = blockIdx.x;
  __shared__ int cnt[NODES * NODES];
  int tid = threadIdx.x;
  if (tid < NODES * NODES) cnt[tid] = 0;
  __syncthreads();
  if (tid < EPG) {
    int s = src[(size_t)g * EPG + tid] - g * NODES;
    int d = dst[(size_t)g * EPG + tid] - g * NODES;
    atomicAdd(&cnt[d * NODES + s], 1);
  }
  __syncthreads();
  if (tid < NODES * NODES) {
    int d = tid / NODES, s = tid % NODES;
    float od = 0.f, id = 0.f;
#pragma unroll
    for (int k = 0; k < NODES; ++k) { od += (float)cnt[k * NODES + s];
                                      id += (float)cnt[d * NODES + k]; }
    od = fmaxf(od, 1.0f); id = fmaxf(id, 1.0f);
    Mg[(size_t)g * 121 + tid] = (float)cnt[tid] * rsqrtf(od) * rsqrtf(id);
  }
}

// ---------------------------------------------------------------------------
// LSTM head
// ---------------------------------------------------------------------------
__global__ void ff1_k(const float* __restrict__ h, const float* __restrict__ Wf1,
                      const float* __restrict__ bf1, float* __restrict__ t64) {
  size_t i = (size_t)blockIdx.x * blockDim.x + threadIdx.x;   // BG*64
  int g = (int)(i >> 6), o = (int)(i & 63);
  float acc = bf1[o];
  const float* hr = h + (size_t)g * HH;
  for (int k = 0; k < HH; ++k) acc += hr[k] * Wf1[(size_t)k * 64 + o];
  t64[i] = acc;
}
__global__ void ff2_k(const float* __restrict__ t64, const float* __restrict__ Wf2,
                      const float* __restrict__ bf2, float* __restrict__ out2) {
  size_t i = (size_t)blockIdx.x * blockDim.x + threadIdx.x;   // BG*2
  int g = (int)(i >> 1), o = (int)(i & 1);
  float acc = bf2[o];
  const float* tr = t64 + (size_t)g * 64;
  for (int k = 0; k < 64; ++k) acc += tr[k] * Wf2[k * 2 + o];
  out2[i] = acc;
}

// ---------------------------------------------------------------------------
// GraphConv aggregation (in-place, bf16, F=256): h[d] = act(M @ x + b)
// Stages the contiguous 11x256 bf16 block via CDNA5 async global->LDS.
// ---------------------------------------------------------------------------
__global__ __launch_bounds__(256)
void agg256(unsigned short* __restrict__ X, const float* __restrict__ Mg,
            const float* __restrict__ bias, int act /*0=relu 1=tanh*/) {
  const int g = blockIdx.x, f = threadIdx.x;
  __shared__ float Ms[NODES * NODES];
  __shared__ alignas(16) unsigned short Xs[NODES * 256];   // 5632 B contiguous
  const unsigned short* base = X + (size_t)g * NODES * 256;
#if HAVE_ASYNC_LDS
  for (int c = threadIdx.x; c < (NODES * 256) / 8; c += blockDim.x) {
    __builtin_amdgcn_global_load_async_to_lds_b128(
        (AS1 i32x4*)(void*)(base + c * 8),
        (AS3 i32x4*)(void*)(&Xs[c * 8]), 0, 0);
  }
  if (f < NODES * NODES) Ms[f] = Mg[(size_t)g * 121 + f];
  __builtin_amdgcn_s_wait_asynccnt(0);
  __syncthreads();
#else
  for (int i = threadIdx.x; i < NODES * 256; i += blockDim.x) Xs[i] = base[i];
  if (f < NODES * NODES) Ms[f] = Mg[(size_t)g * 121 + f];
  __syncthreads();
#endif
  float xv[NODES];
#pragma unroll
  for (int s = 0; s < NODES; ++s) xv[s] = bf2f(Xs[s * 256 + f]);
  float bb = bias[f];
#pragma unroll
  for (int d = 0; d < NODES; ++d) {
    float a = bb;
#pragma unroll
    for (int s = 0; s < NODES; ++s) a += Ms[d * NODES + s] * xv[s];
    a = act ? tanhf(a) : fmaxf(a, 0.0f);
    X[(size_t)g * NODES * 256 + (size_t)d * 256 + f] = f2bf(a);
  }
}

// layer-3 projection (tiny): Xw3[n][o] = sum_k h2bf[n][k] * W3[k][o]
__global__ void gemm3_k(const unsigned short* __restrict__ h2,
                        const float* __restrict__ W3, float* __restrict__ xw3) {
  size_t i = (size_t)blockIdx.x * blockDim.x + threadIdx.x;   // NN*8
  int n = (int)(i >> 3), o = (int)(i & 7);
  float acc = 0.f;
  const unsigned short* hr = h2 + (size_t)n * 256;
  for (int k = 0; k < 256; ++k) acc += bf2f(hr[k]) * W3[(size_t)k * 8 + o];
  xw3[i] = acc;
}

// layer-3 aggregation in-place (F=8)
__global__ void agg8(float* __restrict__ X, const float* __restrict__ Mg,
                     const float* __restrict__ b3) {
  size_t i = (size_t)blockIdx.x * blockDim.x + threadIdx.x;   // BG*8
  int g = (int)(i >> 3), f = (int)(i & 7);
  float xv[NODES];
#pragma unroll
  for (int s = 0; s < NODES; ++s) xv[s] = X[((size_t)(g * NODES + s)) * 8 + f];
  const float* M = Mg + (size_t)g * 121;
  float ov[NODES];
#pragma unroll
  for (int d = 0; d < NODES; ++d) {
    float a = b3[f];
#pragma unroll
    for (int s = 0; s < NODES; ++s) a += M[d * NODES + s] * xv[s];
    ov[d] = fmaxf(a, 0.0f);
  }
#pragma unroll
  for (int d = 0; d < NODES; ++d) X[((size_t)(g * NODES + d)) * 8 + f] = ov[d];
}

// mean_nodes + classify1 + add LSTM head
__global__ void final_k(const float* __restrict__ h3, const float* __restrict__ Wc,
                        const float* __restrict__ bc, const float* __restrict__ out2,
                        float* __restrict__ dout) {
  int g = blockIdx.x * blockDim.x + threadIdx.x;
  if (g >= BG) return;
  float hg[8];
#pragma unroll
  for (int j = 0; j < 8; ++j) {
    float a = 0.f;
#pragma unroll
    for (int d = 0; d < NODES; ++d) a += h3[((size_t)(g * NODES + d)) * 8 + j];
    hg[j] = a * (1.0f / NODES);
  }
#pragma unroll
  for (int c = 0; c < 2; ++c) {
    float y = bc[c];
#pragma unroll
    for (int j = 0; j < 8; ++j) y += hg[j] * Wc[j * 2 + c];
    dout[(size_t)g * 2 + c] = y + out2[(size_t)g * 2 + c];
  }
}

// ---------------------------------------------------------------------------
// Launch
// ---------------------------------------------------------------------------
extern "C" void kernel_launch(void* const* d_in, const int* in_sizes, int n_in,
                              void* d_out, int out_size, void* d_ws, size_t ws_size,
                              hipStream_t stream) {
  const float* feature = (const float*)d_in[0];
  const int*   src     = (const int*)d_in[1];
  const int*   dst     = (const int*)d_in[2];
  const float* W1  = (const float*)d_in[3];
  const float* b1  = (const float*)d_in[4];
  const float* W2  = (const float*)d_in[5];
  const float* b2  = (const float*)d_in[6];
  const float* W3  = (const float*)d_in[7];
  const float* b3  = (const float*)d_in[8];
  const float* Wih = (const float*)d_in[9];
  const float* Whh = (const float*)d_in[10];
  const float* bih = (const float*)d_in[11];
  const float* bhh = (const float*)d_in[12];
  const float* Wf1 = (const float*)d_in[13];
  const float* bf1 = (const float*)d_in[14];
  const float* Wf2 = (const float*)d_in[15];
  const float* bf2 = (const float*)d_in[16];
  const float* Wc  = (const float*)d_in[17];
  const float* bcc = (const float*)d_in[18];
  float* dout = (float*)d_out;

  // ---- workspace layout (bump allocator, 256B aligned) ----
  char* ws = (char*)d_ws;
  size_t off = 0;
  auto alloc = [&](size_t bytes) { size_t r = off; off = (off + bytes + 255) & ~(size_t)255; return r; };
  size_t o_featbf = alloc((size_t)NN * IND * 2);
  size_t o_w1t    = alloc((size_t)256 * 256 * 2);
  size_t o_w2t    = alloc((size_t)256 * 256 * 2);
  size_t o_wcomb  = alloc((size_t)G4 * 512 * 2);
  size_t o_bcomb  = alloc((size_t)G4 * 4);
  size_t o_Mg     = alloc((size_t)BG * 121 * 4);
  size_t o_xw3    = alloc((size_t)NN * 8 * 4);
  size_t o_out2   = alloc((size_t)BG * 2 * 4);
  size_t o_big    = off;
  // LSTM-phase carve-out of the big region (ping-pong bf16 h, c, f32 h, ff tmp):
  size_t o_hbf0 = o_big;
  size_t o_hbf1 = o_hbf0 + (size_t)BG * HH * 2;
  size_t o_cb   = o_hbf1 + (size_t)BG * HH * 2;
  size_t o_hf   = o_cb   + (size_t)BG * HH * 4;
  size_t o_t64  = o_hf   + (size_t)BG * HH * 4;
  // GConv-phase reuse of the big region:
  size_t o_b1 = o_big;
  size_t o_b2 = o_big + (size_t)NN * 256 * 2;

  unsigned short* featbf = (unsigned short*)(ws + o_featbf);
  unsigned short* w1t    = (unsigned short*)(ws + o_w1t);
  unsigned short* w2t    = (unsigned short*)(ws + o_w2t);
  unsigned short* wcomb  = (unsigned short*)(ws + o_wcomb);
  float*          bcomb  = (float*)(ws + o_bcomb);
  float*          Mg     = (float*)(ws + o_Mg);
  float*          xw3    = (float*)(ws + o_xw3);
  float*          out2   = (float*)(ws + o_out2);
  unsigned short* hbf[2] = { (unsigned short*)(ws + o_hbf0),
                             (unsigned short*)(ws + o_hbf1) };
  float*          cb     = (float*)(ws + o_cb);
  float*          hf32   = (float*)(ws + o_hf);
  float*          t64    = (float*)(ws + o_t64);
  unsigned short* b1buf  = (unsigned short*)(ws + o_b1);
  unsigned short* b2buf  = (unsigned short*)(ws + o_b2);

  auto blks = [](size_t n, int t) { return (unsigned)((n + t - 1) / t); };

  // ---- prep: conversions, combined weights/bias, per-graph adjacency ----
  conv_f32_bf16<<<blks((size_t)NN * IND, 256), 256, 0, stream>>>(featbf, feature, (size_t)NN * IND);
  convT_f32_bf16<<<blks((size_t)256 * 256, 256), 256, 0, stream>>>(w1t, W1, 256, 256);
  convT_f32_bf16<<<blks((size_t)256 * 256, 256), 256, 0, stream>>>(w2t, W2, 256, 256);
  conv_rows_bf16<<<blks((size_t)G4 * 256, 256), 256, 0, stream>>>(wcomb, Wih, G4, 256, 512, 0);
  conv_rows_bf16<<<blks((size_t)G4 * 256, 256), 256, 0, stream>>>(wcomb, Whh, G4, 256, 512, 256);
  bias_comb_k<<<blks(G4, 256), 256, 0, stream>>>(bcomb, bih, bhh, G4);
  graph_prep<<<BG, 128, 0, stream>>>(src, dst, Mg);

  // ---- LSTM branch (fused WMMA + gates, h ping-pong, c in place) ----
  zero_u16<<<blks((size_t)BG * HH, 256), 256, 0, stream>>>(hbf[0], (size_t)BG * HH);
  zero_f32<<<blks((size_t)BG * HH, 256), 256, 0, stream>>>(cb, (size_t)BG * HH);

  const unsigned lstm_blocks = ((BG / 16) * (HH / 16) + 7) / 8;   // 16384 waves
  for (int t = 0; t < NODES; ++t) {
    lstm_step_wmma<<<lstm_blocks, 256, 0, stream>>>(
        featbf, hbf[t & 1], hbf[(t & 1) ^ 1], wcomb, bcomb, cb, hf32, t);
  }
  ff1_k<<<blks((size_t)BG * 64, 256), 256, 0, stream>>>(hf32, Wf1, bf1, t64);
  ff2_k<<<blks((size_t)BG * 2, 128), 128, 0, stream>>>(t64, Wf2, bf2, out2);

  // ---- GraphConv branch (big region now free for activations) ----
  const unsigned gemmG_blocks = ((NN / 16) * (256 / 64) + 7) / 8;  // 45056 waves
  wmma_gemm_bf16<<<gemmG_blocks, 256, 0, stream>>>(featbf, w1t, nullptr, b1buf,
                                                   NN, 256, 256);
  agg256<<<BG, 256, 0, stream>>>(b1buf, Mg, b1, 0);                // relu
  wmma_gemm_bf16<<<gemmG_blocks, 256, 0, stream>>>(b1buf, w2t, nullptr, b2buf,
                                                   NN, 256, 256);
  agg256<<<BG, 256, 0, stream>>>(b2buf, Mg, b2, 1);                // tanh
  gemm3_k<<<blks((size_t)NN * 8, 256), 256, 0, stream>>>(b2buf, W3, xw3);
  agg8<<<blks((size_t)BG * 8, 256), 256, 0, stream>>>(xw3, Mg, b3);

  // ---- pooling + classify + add heads ----
  final_k<<<blks(BG, 128), 128, 0, stream>>>(xw3, Wc, bcc, out2, dout);
  (void)in_sizes; (void)n_in; (void)out_size; (void)ws_size;
}
